// MultiViewFusion_23356032155828
// MI455X (gfx1250) — matrix operand order repaired and checked by reading, not compiled, not run on hardware
//
#include <hip/hip_runtime.h>
#include <stdint.h>

#define NROWS 65536
#define TDIM  768
#define UDIM  256
#define EDIM  128
#define ODIM  256

typedef __attribute__((ext_vector_type(16))) __bf16 v16bf;
typedef __attribute__((ext_vector_type(8)))  float  v8f;

union V16 { v16bf v; uint4 q[2]; unsigned short s[16]; };

__device__ __forceinline__ unsigned short f2bf(float f) {
    unsigned u = __float_as_uint(f);
    u += 0x7FFFu + ((u >> 16) & 1u);          // round-to-nearest-even
    return (unsigned short)(u >> 16);
}
__device__ __forceinline__ float bf2f(unsigned short s) {
    return __uint_as_float(((unsigned)s) << 16);
}

__device__ __forceinline__ v8f wmma_bf16(v16bf a, v16bf b, v8f c) {
    return __builtin_amdgcn_wmma_f32_16x16x32_bf16(false, a, false, b, (short)0, c, false, false);
}

// A fragment (16x32 bf16, row-major src, stride ld elements).
// lane 0-15: row=l16, K = k0+0..7 and k0+16..23 ; lanes 16-31: +8 on both halves
__device__ __forceinline__ v16bf afrag_bf16(const unsigned short* base, int ld, int row,
                                            int k0, int hf) {
    const unsigned short* p = base + (size_t)row * ld + k0 + hf * 8;
    V16 r;
    r.q[0] = *(const uint4*)p;
    r.q[1] = *(const uint4*)(p + 16);
    return r.v;
}

// A fragment from f32 memory with on-the-fly bf16 conversion
__device__ __forceinline__ v16bf afrag_f32(const float* __restrict__ A, int ld, int row,
                                           int k0, int hf) {
    const float* p = A + (size_t)row * ld + k0 + hf * 8;
    float4 f0 = *(const float4*)(p);
    float4 f1 = *(const float4*)(p + 4);
    float4 f2 = *(const float4*)(p + 16);
    float4 f3 = *(const float4*)(p + 20);
    V16 r;
    r.s[0]=f2bf(f0.x); r.s[1]=f2bf(f0.y); r.s[2]=f2bf(f0.z); r.s[3]=f2bf(f0.w);
    r.s[4]=f2bf(f1.x); r.s[5]=f2bf(f1.y); r.s[6]=f2bf(f1.z); r.s[7]=f2bf(f1.w);
    r.s[8]=f2bf(f2.x); r.s[9]=f2bf(f2.y); r.s[10]=f2bf(f2.z); r.s[11]=f2bf(f2.w);
    r.s[12]=f2bf(f3.x); r.s[13]=f2bf(f3.y); r.s[14]=f2bf(f3.z); r.s[15]=f2bf(f3.w);
    return r.v;
}

// B fragment for W.T where W is [N,K] row-major bf16: lane needs W[n, k0+hf*8 .. +7] and W[n, k0+16+hf*8 .. +7]
__device__ __forceinline__ v16bf bfrag_bf16(const unsigned short* __restrict__ W, int ld,
                                            int n, int k0, int hf) {
    const unsigned short* p = W + (size_t)n * ld + k0 + hf * 8;
    V16 r;
    r.q[0] = *(const uint4*)p;
    r.q[1] = *(const uint4*)(p + 16);
    return r.v;
}

// ---------------- setup kernels ----------------

__global__ void k_cvt(const float* __restrict__ src, unsigned short* __restrict__ dst, int n) {
    int i = blockIdx.x * blockDim.x + threadIdx.x;
    if (i < n) dst[i] = f2bf(src[i]);
}

// Wc[i] = opw[i] @ wv[i]  (wv = ipw[i][2E:3E])  -> bf16 [4][128][128]
__global__ void k_combine_w(const float* __restrict__ ipw, const float* __restrict__ opw,
                            unsigned short* __restrict__ Wc) {
    int i = blockIdx.x >> 7;
    int o = blockIdx.x & 127;
    int k = threadIdx.x;
    const float* op = opw + ((size_t)i * EDIM + o) * EDIM;
    const float* wv = ipw + (size_t)i * 3 * EDIM * EDIM + (size_t)2 * EDIM * EDIM + k;
    float acc = 0.f;
    for (int j = 0; j < EDIM; ++j) acc += op[j] * wv[(size_t)j * EDIM];
    Wc[((size_t)i * EDIM + o) * EDIM + k] = f2bf(acc);
}

// bc[i] = opw[i] @ bv[i] + opb[i]
__global__ void k_combine_b(const float* __restrict__ ipb, const float* __restrict__ opw,
                            const float* __restrict__ opb, float* __restrict__ bc) {
    int i = blockIdx.x;
    int o = threadIdx.x;
    const float* op = opw + ((size_t)i * EDIM + o) * EDIM;
    const float* bv = ipb + (size_t)i * 3 * EDIM + 2 * EDIM;
    float acc = opb[(size_t)i * EDIM + o];
    for (int j = 0; j < EDIM; ++j) acc += op[j] * bv[j];
    bc[i * EDIM + o] = acc;
}

// ---------------- stage 1: input modality projections ----------------
// out[B,128] = bf16( A[B,K] @ W[128,K].T + bias )
template <int K>
__global__ __launch_bounds__(256) void k_proj(const float* __restrict__ A,
                                              const unsigned short* __restrict__ W,
                                              const float* __restrict__ bias,
                                              unsigned short* __restrict__ out) {
    int tid = threadIdx.x;
    int wave = tid >> 5, lane = tid & 31, hf = lane >> 4, l16 = lane & 15;
    int rowBase = blockIdx.x * 128 + wave * 16;
    v8f acc[8] = {};
    for (int k0 = 0; k0 < K; k0 += 32) {
        v16bf a = afrag_f32(A, K, rowBase + l16, k0, hf);
#pragma unroll
        for (int nt = 0; nt < 8; ++nt) {
            v16bf b = bfrag_bf16(W, K, nt * 16 + l16, k0, hf);
            acc[nt] = wmma_bf16(a, b, acc[nt]);
        }
    }
#pragma unroll
    for (int nt = 0; nt < 8; ++nt) {
        float bv = bias[nt * 16 + l16];
#pragma unroll
        for (int r = 0; r < 8; ++r) {
            int row = rowBase + hf * 8 + r;
            out[(size_t)row * EDIM + nt * 16 + l16] = f2bf(acc[nt][r] + bv);
        }
    }
}

// ---------------- stage 2: fused projections + gate + blend ----------------

#define SCR_LD 264   // 16x(256+8) bf16 scratch per wave, padded to spread LDS banks

__device__ __forceinline__ void phase(const unsigned short* __restrict__ mSelf,
                                      const unsigned short* __restrict__ mCross,
                                      const unsigned short* __restrict__ Wself,
                                      const unsigned short* __restrict__ Wcross,
                                      const float* __restrict__ bSelf,
                                      const float* __restrict__ bCross,
                                      const unsigned short* __restrict__ gW,
                                      const float* __restrict__ gb,
                                      unsigned short* scr,
                                      unsigned short* __restrict__ fusedOut, int colOff,
                                      int rowBase, int hf, int l16) {
    v8f accS[8] = {};
    v8f accC[8] = {};
    for (int k0 = 0; k0 < 128; k0 += 32) {
        v16bf aS = afrag_bf16(mSelf, EDIM, rowBase + l16, k0, hf);
        v16bf aC = afrag_bf16(mCross, EDIM, rowBase + l16, k0, hf);
#pragma unroll
        for (int nt = 0; nt < 8; ++nt) {
            v16bf b1 = bfrag_bf16(Wself, EDIM, nt * 16 + l16, k0, hf);
            accS[nt] = wmma_bf16(aS, b1, accS[nt]);
            v16bf b2 = bfrag_bf16(Wcross, EDIM, nt * 16 + l16, k0, hf);
            accC[nt] = wmma_bf16(aC, b2, accC[nt]);
        }
    }
    // bias + stage [self | cross] into LDS scratch (D-layout -> row-major 16x256)
#pragma unroll
    for (int nt = 0; nt < 8; ++nt) {
        float bs = bSelf[nt * 16 + l16];
        float bx = bCross[nt * 16 + l16];
#pragma unroll
        for (int r = 0; r < 8; ++r) {
            int rr = hf * 8 + r, cc = nt * 16 + l16;
            scr[rr * SCR_LD + cc]       = f2bf(accS[nt][r] + bs);
            scr[rr * SCR_LD + 128 + cc] = f2bf(accC[nt][r] + bx);
        }
    }
    __syncthreads();
    // gate logits: [self|cross] @ gW.T   (K = 256)
    v8f accG[8] = {};
    for (int k0 = 0; k0 < 256; k0 += 32) {
        v16bf a = afrag_bf16(scr, SCR_LD, l16, k0, hf);
#pragma unroll
        for (int nt = 0; nt < 8; ++nt) {
            v16bf b = bfrag_bf16(gW, 256, nt * 16 + l16, k0, hf);
            accG[nt] = wmma_bf16(a, b, accG[nt]);
        }
    }
    // sigmoid gate + blend + store to fused
#pragma unroll
    for (int nt = 0; nt < 8; ++nt) {
        float gbv = gb[nt * 16 + l16];
#pragma unroll
        for (int r = 0; r < 8; ++r) {
            int rr = hf * 8 + r, cc = nt * 16 + l16;
            float s = bf2f(scr[rr * SCR_LD + cc]);
            float c = bf2f(scr[rr * SCR_LD + 128 + cc]);
            float g = 1.0f / (1.0f + __expf(-(accG[nt][r] + gbv)));
            float fv = s + g * (c - s);
            fusedOut[(size_t)(rowBase + rr) * 256 + colOff + cc] = f2bf(fv);
        }
    }
    __syncthreads();   // scratch reused by next phase
}

__global__ __launch_bounds__(128) void k_attn_gate(const unsigned short* __restrict__ textm,
                                                   const unsigned short* __restrict__ userm,
                                                   const unsigned short* __restrict__ Wc,
                                                   const float* __restrict__ bc,
                                                   const unsigned short* __restrict__ tgW,
                                                   const float* __restrict__ tgb,
                                                   const unsigned short* __restrict__ ugW,
                                                   const float* __restrict__ ugb,
                                                   unsigned short* __restrict__ fused) {
    __shared__ unsigned short scr[4][16 * SCR_LD];
    int tid = threadIdx.x;
    int wave = tid >> 5, lane = tid & 31, hf = lane >> 4, l16 = lane & 15;
    int rowBase = blockIdx.x * 64 + wave * 16;
    // text:  self = text_m@W0, cross = user_m@W2, gate tg -> fused[:, 0:128]
    phase(textm, userm, Wc + 0 * 16384, Wc + 2 * 16384, bc + 0, bc + 256,
          tgW, tgb, scr[wave], fused, 0, rowBase, hf, l16);
    // user:  self = user_m@W1, cross = text_m@W3, gate ug -> fused[:, 128:256]
    phase(userm, textm, Wc + 1 * 16384, Wc + 3 * 16384, bc + 128, bc + 384,
          ugW, ugb, scr[wave], fused, 128, rowBase, hf, l16);
}

// ---------------- stage 3: ff GEMM + LayerNorm + ReLU ----------------

__global__ __launch_bounds__(256) void k_ff_ln(const unsigned short* __restrict__ fused,
                                               const unsigned short* __restrict__ ffW,
                                               const float* __restrict__ ffb,
                                               const float* __restrict__ lng,
                                               const float* __restrict__ lnb,
                                               float* __restrict__ out) {
    int tid = threadIdx.x;
    int wave = tid >> 5, lane = tid & 31, hf = lane >> 4, l16 = lane & 15;
    int rowBase = blockIdx.x * 128 + wave * 16;
    v8f acc[16] = {};
    for (int k0 = 0; k0 < 256; k0 += 32) {
        v16bf a = afrag_bf16(fused, 256, rowBase + l16, k0, hf);
#pragma unroll
        for (int nt = 0; nt < 16; ++nt) {
            v16bf b = bfrag_bf16(ffW, 256, nt * 16 + l16, k0, hf);
            acc[nt] = wmma_bf16(a, b, acc[nt]);
        }
    }
    float s[8] = {}, sq[8] = {};
#pragma unroll
    for (int nt = 0; nt < 16; ++nt) {
        float bb = ffb[nt * 16 + l16];
#pragma unroll
        for (int r = 0; r < 8; ++r) {
            float h = acc[nt][r] + bb;
            acc[nt][r] = h;
            s[r] += h;
            sq[r] += h * h;
        }
    }
    // reduce across the 16 lanes of each half (xor 8,4,2,1 stays within a half in wave32)
#pragma unroll
    for (int m = 8; m >= 1; m >>= 1) {
#pragma unroll
        for (int r = 0; r < 8; ++r) {
            s[r] += __shfl_xor(s[r], m, 32);
            sq[r] += __shfl_xor(sq[r], m, 32);
        }
    }
    float mu[8], rs[8];
#pragma unroll
    for (int r = 0; r < 8; ++r) {
        mu[r] = s[r] * (1.0f / 256.0f);
        float var = sq[r] * (1.0f / 256.0f) - mu[r] * mu[r];
        rs[r] = rsqrtf(var + 1e-5f);
    }
#pragma unroll
    for (int nt = 0; nt < 16; ++nt) {
        float g = lng[nt * 16 + l16], b2 = lnb[nt * 16 + l16];
#pragma unroll
        for (int r = 0; r < 8; ++r) {
            float v = (acc[nt][r] - mu[r]) * rs[r] * g + b2;
            out[(size_t)(rowBase + hf * 8 + r) * 256 + nt * 16 + l16] = fmaxf(v, 0.0f);
        }
    }
}

// ---------------- host launcher ----------------

extern "C" void kernel_launch(void* const* d_in, const int* in_sizes, int n_in,
                              void* d_out, int out_size, void* d_ws, size_t ws_size,
                              hipStream_t stream) {
    const float* text = (const float*)d_in[0];
    const float* user = (const float*)d_in[1];
    const float* tm_w = (const float*)d_in[2];
    const float* tm_b = (const float*)d_in[3];
    const float* um_w = (const float*)d_in[4];
    const float* um_b = (const float*)d_in[5];
    const float* ipw  = (const float*)d_in[6];
    const float* ipb  = (const float*)d_in[7];
    const float* opw  = (const float*)d_in[8];
    const float* opb  = (const float*)d_in[9];
    const float* tg_w = (const float*)d_in[10];
    const float* tg_b = (const float*)d_in[11];
    const float* ug_w = (const float*)d_in[12];
    const float* ug_b = (const float*)d_in[13];
    const float* ff_w = (const float*)d_in[14];
    const float* ff_b = (const float*)d_in[15];
    const float* ln_g = (const float*)d_in[16];
    const float* ln_b = (const float*)d_in[17];

    char* ws = (char*)d_ws;
    size_t off = 0;
    auto alloc = [&](size_t bytes) {
        size_t o = off;
        off += (bytes + 255) & ~(size_t)255;
        return o;
    };
    unsigned short* Wc    = (unsigned short*)(ws + alloc((size_t)4 * 128 * 128 * 2));
    float*          bcomb = (float*)(ws + alloc((size_t)4 * 128 * 4));
    unsigned short* tmW   = (unsigned short*)(ws + alloc((size_t)128 * 768 * 2));
    unsigned short* umW   = (unsigned short*)(ws + alloc((size_t)128 * 256 * 2));
    unsigned short* tgW   = (unsigned short*)(ws + alloc((size_t)128 * 256 * 2));
    unsigned short* ugW   = (unsigned short*)(ws + alloc((size_t)128 * 256 * 2));
    unsigned short* ffW   = (unsigned short*)(ws + alloc((size_t)256 * 256 * 2));
    unsigned short* textm = (unsigned short*)(ws + alloc((size_t)NROWS * 128 * 2));
    unsigned short* userm = (unsigned short*)(ws + alloc((size_t)NROWS * 128 * 2));
    unsigned short* fused = (unsigned short*)(ws + alloc((size_t)NROWS * 256 * 2));
    (void)ws_size; (void)in_sizes; (void)n_in; (void)out_size;

    // setup: fold (v-proj, out-proj) into one matrix each; convert weights to bf16
    k_combine_w<<<4 * 128, 128, 0, stream>>>(ipw, opw, Wc);
    k_combine_b<<<4, 128, 0, stream>>>(ipb, opw, opb, bcomb);
    k_cvt<<<(128 * 768 + 255) / 256, 256, 0, stream>>>(tm_w, tmW, 128 * 768);
    k_cvt<<<(128 * 256 + 255) / 256, 256, 0, stream>>>(um_w, umW, 128 * 256);
    k_cvt<<<(128 * 256 + 255) / 256, 256, 0, stream>>>(tg_w, tgW, 128 * 256);
    k_cvt<<<(128 * 256 + 255) / 256, 256, 0, stream>>>(ug_w, ugW, 128 * 256);
    k_cvt<<<(256 * 256 + 255) / 256, 256, 0, stream>>>(ff_w, ffW, 256 * 256);

    // stage 1: modality projections (WMMA, f32 inputs converted on the fly)
    k_proj<TDIM><<<NROWS / 128, 256, 0, stream>>>(text, tmW, tm_b, textm);
    k_proj<UDIM><<<NROWS / 128, 256, 0, stream>>>(user, umW, um_b, userm);

    // stage 2: four fused attention projections + gates + blend
    k_attn_gate<<<NROWS / 64, 128, 0, stream>>>(textm, userm, Wc, bcomb,
                                                tgW, tg_b, ugW, ug_b, fused);

    // stage 3: ff GEMM + LayerNorm + ReLU
    k_ff_ln<<<NROWS / 128, 256, 0, stream>>>(fused, ffW, ff_b, ln_g, ln_b, (float*)d_out);
}
